// Qwen3MoeSparseMoeBlock_42262478192981
// MI455X (gfx1250) — compile-verified, matrix operand order checked by reading, not run on hardware
//
#include <hip/hip_runtime.h>

#define T_TOK 1024
#define HID   2048
#define NEXP  64
#define TOPK  8
#define INTER 768

typedef __attribute__((ext_vector_type(16))) _Float16 v16h;
typedef __attribute__((ext_vector_type(8)))  _Float16 v8h;
typedef __attribute__((ext_vector_type(4)))  _Float16 v4h;
typedef __attribute__((ext_vector_type(2)))  _Float16 v2h;
typedef __attribute__((ext_vector_type(8)))  float    v8f;

#define LDA 40   // A tile leading dim (halves): 80B rows, conflict-free, 8B-aligned v4h
#define LDB 36   // B^T tile leading dim (halves): 72B rows, conflict-free, 8B-aligned v4h

// ---------------- fragment loader (ISA 7.12.2 16-bit layouts, wave32) ----------------
// Row-major [row][k] tile with K-chunk 32 contiguous per row.
// lane<16 -> row=lane, elems[0..7]=K0..7, [8..15]=K16..23
// lane>=16 -> row=lane-16, elems[0..7]=K8..15, [8..15]=K24..31
__device__ __forceinline__ v16h frag_rowk(const _Float16* S, int ld, int row16, int lane) {
  const int r = lane & 15, hi = lane >> 4;
  const _Float16* base = S + (row16 + r) * ld;
  v4h a0 = *(const v4h*)(base + hi * 8);
  v4h a1 = *(const v4h*)(base + hi * 8 + 4);
  v4h a2 = *(const v4h*)(base + 16 + hi * 8);
  v4h a3 = *(const v4h*)(base + 16 + hi * 8 + 4);
  v8h lo = __builtin_shufflevector(a0, a1, 0,1,2,3,4,5,6,7);
  v8h hh = __builtin_shufflevector(a2, a3, 0,1,2,3,4,5,6,7);
  return __builtin_shufflevector(lo, hh, 0,1,2,3,4,5,6,7,8,9,10,11,12,13,14,15);
}

// ---------------- 1) zero the output (gemm2 accumulates atomically) ----------------
__global__ void zero_out_kernel(float* __restrict__ out, int n4) {
  float4 z = {0.f, 0.f, 0.f, 0.f};
  for (int i = blockIdx.x * blockDim.x + threadIdx.x; i < n4; i += gridDim.x * blockDim.x)
    ((float4*)out)[i] = z;
}

// ---------------- 2) router: softmax -> top-8 -> renormalize ----------------
__global__ void router_kernel(const float* __restrict__ x, const float* __restrict__ gw,
                              int* __restrict__ topk_idx, float* __restrict__ topk_w) {
  __shared__ float xs[HID];
  __shared__ float probs[NEXP];
  const int t = blockIdx.x;
  const int tid = threadIdx.x;                 // 64 threads = 1 expert each
  for (int i = tid; i < HID; i += 64) xs[i] = x[(size_t)t * HID + i];
  __syncthreads();
  float acc = 0.f;
  for (int h = 0; h < HID; ++h) acc += xs[h] * gw[(size_t)h * NEXP + tid];
  probs[tid] = acc;
  __syncthreads();
  float mx = -1e30f;
  for (int j = 0; j < NEXP; ++j) mx = fmaxf(mx, probs[j]);
  float p = __expf(acc - mx);
  __syncthreads();
  probs[tid] = p;
  __syncthreads();
  float sum = 0.f;
  for (int j = 0; j < NEXP; ++j) sum += probs[j];
  __syncthreads();
  probs[tid] = p / sum;
  __syncthreads();
  if (tid == 0) {
    float vals[TOPK]; int idxs[TOPK]; float s = 0.f;
    for (int k = 0; k < TOPK; ++k) {
      float best = -1.f; int bi = 0;
      for (int j = 0; j < NEXP; ++j)
        if (probs[j] > best) { best = probs[j]; bi = j; }
      vals[k] = best; idxs[k] = bi; probs[bi] = -2.f; s += best;
    }
    for (int k = 0; k < TOPK; ++k) {
      topk_idx[t * TOPK + k] = idxs[k];
      topk_w[t * TOPK + k]   = vals[k] / s;
    }
  }
}

// ---------------- 3) deterministic per-expert token lists (1 wave32 / expert) ----------------
__global__ void build_lists_kernel(const int* __restrict__ topk_idx,
                                   const float* __restrict__ topk_w,
                                   int* __restrict__ tok_list, float* __restrict__ w_list,
                                   int* __restrict__ counts) {
  const int e = blockIdx.x;
  const int lane = threadIdx.x;                // 32 threads
  int cnt = 0;
  for (int t0 = 0; t0 < T_TOK; t0 += 32) {
    const int t = t0 + lane;
    float w = 0.f; bool pred = false;
#pragma unroll
    for (int k = 0; k < TOPK; ++k) {
      if (topk_idx[t * TOPK + k] == e) { pred = true; w = topk_w[t * TOPK + k]; }
    }
    unsigned mask = (unsigned)__ballot(pred);  // wave32 -> low 32 bits
    if (pred) {
      int pos = cnt + __popc(mask & ((1u << lane) - 1u));
      tok_list[e * T_TOK + pos] = t;
      w_list[e * T_TOK + pos]   = w;
    }
    cnt += __popc(mask);
  }
  if (lane == 0) counts[e] = cnt;
}

// ---------------- 4) exclusive scan of expert counts ----------------
__global__ void scan_offsets_kernel(const int* __restrict__ counts, int* __restrict__ offs) {
  if (blockIdx.x == 0 && threadIdx.x == 0) {
    int s = 0;
    for (int e = 0; e < NEXP; ++e) { offs[e] = s; s += counts[e]; }
    offs[NEXP] = s;
  }
}

// ---------------- 5) grouped GEMM1: g = X@Wg, u = X@Wu, h = silu(g)*u*rw ----------------
// block = 256 thr = 8 waves; tile M=32 tokens x N=64 inter cols; computes g AND u.
// Software-pipelined: next chunk's fp32 held in registers while current chunk is WMMA'd.
__global__ void moe_gemm1_kernel(const float* __restrict__ x,
                                 const float* __restrict__ wg_all,
                                 const float* __restrict__ wu_all,
                                 const int* __restrict__ tok_list,
                                 const float* __restrict__ w_list,
                                 const int* __restrict__ counts,
                                 const int* __restrict__ offs,
                                 _Float16* __restrict__ hbuf) {
  const int NT = INTER / 64;                   // 12
  const int MT = T_TOK / 32;                   // 32
  const int b  = blockIdx.x;
  const int e  = b / (MT * NT);
  const int r2 = b - e * (MT * NT);
  const int mt = r2 / NT, nt = r2 % NT;
  const int cnt = counts[e];
  const int m0  = mt * 32;
  if (m0 >= cnt) return;                       // uniform early-exit (static grid)
  const int n0   = nt * 64;
  int rows = cnt - m0; if (rows > 32) rows = 32;

  __shared__ _Float16 As[32 * LDA];            // [m][k]
  __shared__ _Float16 Bgs[64 * LDB];           // [n][k]  (transposed)
  __shared__ _Float16 Bus[64 * LDB];           // [n][k]
  __shared__ int   sTok[32];
  __shared__ float sW[32];

  const int tid  = threadIdx.x;
  const int lane = tid & 31, wid = tid >> 5;
  const int mi = wid >> 2, ni = wid & 3;       // 2 x 4 wave grid

  if (tid < 32) {
    const int mm = (tid < rows) ? tid : rows - 1;
    sTok[tid] = tok_list[e * T_TOK + m0 + mm];
    sW[tid]   = (tid < rows) ? w_list[e * T_TOK + m0 + tid] : 0.f;
  }
  __syncthreads();                             // sTok visible; enables hoisting below

  const float* wg = wg_all + (size_t)e * HID * INTER;
  const float* wu = wu_all + (size_t)e * HID * INTER;

  // A staging map: 256 thr cover 32m x 32k (float4 each); gathered row base hoisted
  const int am = tid >> 3, aq = tid & 7;
  const float* aSrc = x + (size_t)sTok[am] * HID + aq * 4;
  // B staging map: tid<128 -> Wg, tid>=128 -> Wu; 4k x 4n micro-tile, transposed on store
  const float*  bsrc = (tid < 128) ? wg : wu;
  _Float16*     bdst = (tid < 128) ? Bgs : Bus;
  const int bm = tid & 127;
  const int bk = (bm >> 4) << 2;               // 0,4,...,28
  const int bn = (bm & 15) << 2;               // 0,4,...,60
  const float* bp = bsrc + (size_t)bk * INTER + n0 + bn;

  v8f accG = {}; v8f accU = {};

  // pipeline prologue: chunk 0 into registers
  float4 fA = *(const float4*)(aSrc);
  float4 f0 = *(const float4*)(bp);
  float4 f1 = *(const float4*)(bp + INTER);
  float4 f2 = *(const float4*)(bp + 2 * INTER);
  float4 f3 = *(const float4*)(bp + 3 * INTER);

  for (int k0 = 0; k0 < HID; k0 += 32) {
    { // commit staged registers to LDS (fp32 -> f16, packed stores)
      v4h av = {(_Float16)fA.x, (_Float16)fA.y, (_Float16)fA.z, (_Float16)fA.w};
      *(v4h*)(As + am * LDA + aq * 4) = av;
      v4h c0 = {(_Float16)f0.x, (_Float16)f1.x, (_Float16)f2.x, (_Float16)f3.x};
      v4h c1 = {(_Float16)f0.y, (_Float16)f1.y, (_Float16)f2.y, (_Float16)f3.y};
      v4h c2 = {(_Float16)f0.z, (_Float16)f1.z, (_Float16)f2.z, (_Float16)f3.z};
      v4h c3 = {(_Float16)f0.w, (_Float16)f1.w, (_Float16)f2.w, (_Float16)f3.w};
      _Float16* d = bdst + bn * LDB + bk;
      *(v4h*)(d)           = c0;
      *(v4h*)(d + LDB)     = c1;
      *(v4h*)(d + 2 * LDB) = c2;
      *(v4h*)(d + 3 * LDB) = c3;
    }
    __syncthreads();                           // chunk visible to all waves
    if (k0 + 32 < HID) {                       // issue next chunk loads (overlap WMMA)
      const float* np = bp + (size_t)(k0 + 32) * INTER;
      fA = *(const float4*)(aSrc + k0 + 32);
      f0 = *(const float4*)(np);
      f1 = *(const float4*)(np + INTER);
      f2 = *(const float4*)(np + 2 * INTER);
      f3 = *(const float4*)(np + 3 * INTER);
      if (k0 + 64 < HID)                       // stream-ahead -> global_prefetch_b8
        __builtin_prefetch(np + (size_t)32 * INTER, 0, 1);
    }
    const v16h a  = frag_rowk(As,  LDA, mi * 16, lane);
    const v16h bg = frag_rowk(Bgs, LDB, ni * 16, lane);
    const v16h bu = frag_rowk(Bus, LDB, ni * 16, lane);
    accG = __builtin_amdgcn_wmma_f32_16x16x32_f16(false, a, false, bg, (short)0, accG, false, false);
    accU = __builtin_amdgcn_wmma_f32_16x16x32_f16(false, a, false, bu, (short)0, accU, false, false);
    __syncthreads();                           // compute done before next overwrite
  }

  // epilogue: h = silu(g)*u*rw, stored f16 at expert-compacted rows
  const int rl = lane & 15, hi = lane >> 4;
  const int hoff = offs[e];
#pragma unroll
  for (int i = 0; i < 8; ++i) {
    const int m = mi * 16 + hi * 8 + i;        // C row mapping (ISA f32 C/D layout)
    if (m0 + m < cnt) {
      const float g = accG[i], u = accU[i];
      const float hv = (g / (1.f + __expf(-g))) * u * sW[m];
      const size_t row = (size_t)(hoff + m0 + m);
      hbuf[row * INTER + n0 + ni * 16 + rl] = (_Float16)hv;
    }
  }
}

// ---------------- 6) grouped GEMM2: out[t] += h_e @ Wd[e] (f32 atomics) ----------------
__global__ void moe_gemm2_kernel(const _Float16* __restrict__ hbuf,
                                 const float* __restrict__ wd_all,
                                 const int* __restrict__ tok_list,
                                 const int* __restrict__ counts,
                                 const int* __restrict__ offs,
                                 float* __restrict__ out) {
  const int NT = HID / 64;                     // 32
  const int MT = T_TOK / 32;                   // 32
  const int b  = blockIdx.x;
  const int e  = b / (MT * NT);
  const int r2 = b - e * (MT * NT);
  const int mt = r2 / NT, nt = r2 % NT;
  const int cnt = counts[e];
  const int m0  = mt * 32;
  if (m0 >= cnt) return;
  const int n0 = nt * 64;

  __shared__ _Float16 Bs[64 * LDB];            // W_down chunk, transposed [n][k]
  __shared__ int sTok[32];

  const int tid  = threadIdx.x;
  const int lane = tid & 31, wid = tid >> 5;
  const int mi = wid >> 2, ni = wid & 3;
  const int rl = lane & 15, hi = lane >> 4;

  if (tid < 32) {
    int mm = m0 + tid; if (mm >= cnt) mm = cnt - 1;
    sTok[tid] = tok_list[e * T_TOK + mm];
  }

  const int hoff = offs[e];
  int rowG = hoff + m0 + mi * 16 + rl;         // h rows contiguous per expert
  if (rowG > T_TOK * TOPK - 1) rowG = T_TOK * TOPK - 1;  // stay inside hbuf
  const _Float16* hrow = hbuf + (size_t)rowG * INTER;

  const float* wd = wd_all + (size_t)e * INTER * HID;

  // B staging map: 2k x 4n micro-tile per thread (256 thr cover 32k x 64n)
  const int bk = (tid >> 4) << 1;              // 0,2,...,30
  const int bn = (tid & 15) << 2;              // 0,4,...,60
  const float* bp = wd + (size_t)bk * HID + n0 + bn;

  v8f acc = {};

  // pipeline prologue
  float4 r0 = *(const float4*)(bp);
  float4 r1 = *(const float4*)(bp + HID);

  for (int k0 = 0; k0 < INTER; k0 += 32) {
    { // commit staged registers: [32 k][64 n] -> B^T [64 n][32 k]
      v2h c0 = {(_Float16)r0.x, (_Float16)r1.x};
      v2h c1 = {(_Float16)r0.y, (_Float16)r1.y};
      v2h c2 = {(_Float16)r0.z, (_Float16)r1.z};
      v2h c3 = {(_Float16)r0.w, (_Float16)r1.w};
      _Float16* d = Bs + bn * LDB + bk;
      *(v2h*)(d)           = c0;
      *(v2h*)(d + LDB)     = c1;
      *(v2h*)(d + 2 * LDB) = c2;
      *(v2h*)(d + 3 * LDB) = c3;
    }
    __syncthreads();
    if (k0 + 32 < INTER) {                     // next chunk loads overlap WMMA
      const float* np = bp + (size_t)(k0 + 32) * HID;
      r0 = *(const float4*)(np);
      r1 = *(const float4*)(np + HID);
      if (k0 + 64 < INTER) __builtin_prefetch(np + (size_t)32 * HID, 0, 1);
    }
    // A frag straight from global f16 h rows (16B-aligned v8h pairs)
    const v8h lo = *(const v8h*)(hrow + k0 + hi * 8);
    const v8h hh = *(const v8h*)(hrow + k0 + 16 + hi * 8);
    const v16h a = __builtin_shufflevector(lo, hh, 0,1,2,3,4,5,6,7,8,9,10,11,12,13,14,15);
    const v16h bf = frag_rowk(Bs, LDB, ni * 16, lane);
    acc = __builtin_amdgcn_wmma_f32_16x16x32_f16(false, a, false, bf, (short)0, acc, false, false);
    __syncthreads();
  }

#pragma unroll
  for (int i = 0; i < 8; ++i) {
    const int m = mi * 16 + hi * 8 + i;
    if (m0 + m < cnt) {
      const int t = sTok[m];
      atomicAdd(out + (size_t)t * HID + n0 + ni * 16 + rl, acc[i]);
    }
  }
}

// ---------------- host launcher ----------------
extern "C" void kernel_launch(void* const* d_in, const int* in_sizes, int n_in,
                              void* d_out, int out_size, void* d_ws, size_t ws_size,
                              hipStream_t stream) {
  (void)in_sizes; (void)n_in; (void)out_size; (void)ws_size;
  const float* x    = (const float*)d_in[0];   // [1024, 2048]
  const float* gw   = (const float*)d_in[1];   // [2048, 64]
  const float* wgat = (const float*)d_in[2];   // [64, 2048, 768]
  const float* wup  = (const float*)d_in[3];   // [64, 2048, 768]
  const float* wdn  = (const float*)d_in[4];   // [64, 768, 2048]
  float* out = (float*)d_out;

  // workspace carve (~13.2 MB total)
  char* ws = (char*)d_ws;
  int*      counts   = (int*)(ws + 0);                        // 256 B slot
  int*      offs     = (int*)(ws + 256);                      // 65 ints
  int*      topk_idx = (int*)(ws + 1024);                     // 32 KB
  float*    topk_w   = (float*)(ws + 1024 + 32768);           // 32 KB
  int*      tok_list = (int*)(ws + 1024 + 65536);             // 256 KB
  float*    w_list   = (float*)(ws + 1024 + 65536 + 262144);  // 256 KB
  _Float16* hbuf     = (_Float16*)(ws + 1024 + 65536 + 524288); // 8192 x 768 f16 = 12 MB

  zero_out_kernel<<<1024, 256, 0, stream>>>(out, (T_TOK * HID) / 4);
  router_kernel<<<T_TOK, 64, 0, stream>>>(x, gw, topk_idx, topk_w);
  build_lists_kernel<<<NEXP, 32, 0, stream>>>(topk_idx, topk_w, tok_list, w_list, counts);
  scan_offsets_kernel<<<1, 1, 0, stream>>>(counts, offs);
  moe_gemm1_kernel<<<NEXP * (T_TOK / 32) * (INTER / 64), 256, 0, stream>>>(
      x, wgat, wup, tok_list, w_list, counts, offs, hbuf);
  moe_gemm2_kernel<<<NEXP * (T_TOK / 32) * (HID / 64), 256, 0, stream>>>(
      hbuf, wdn, tok_list, counts, offs, out);
}
